// VariationalSDEModel_85787676770669
// MI455X (gfx1250) — compile-verified, matrix operand order checked by reading, not run on hardware
//
#include <hip/hip_runtime.h>
#include <hip/hip_bf16.h>
#include <math.h>

typedef _Float16 h16 __attribute__((ext_vector_type(16)));
typedef float    f8  __attribute__((ext_vector_type(8)));

#define Nn      262144
#define HIDN    128
#define NST     50
#define WAVES   4
#define SPW     16            // samples per wave (WMMA M)
#define SPB     (WAVES*SPW)   // 64 samples per block

// ---- WMMA fragment helpers (ISA 7.12.2 layouts, wave32) -------------------

__device__ __forceinline__ h16 ld16h(const _Float16* p0, const _Float16* p1) {
    union { uint4 u[2]; h16 v; } t;
    t.u[0] = *(const uint4*)p0;
    t.u[1] = *(const uint4*)p1;
    return t.v;
}

// A: 16x32 f16 tile from raw LDS [m*128 + k]; lane(hi=lane>>4,lo=lane&15):
//   elem e<8 : k = kk*32 + hi*8 + e        (contiguous -> one b128)
//   elem e>=8: k = kk*32 + 16 + hi*8 + e-8 (contiguous -> one b128)
__device__ __forceinline__ void loadA(const _Float16* h1, int lo, int hi, h16 A[4]) {
#pragma unroll
    for (int kk = 0; kk < 4; ++kk) {
        const _Float16* base = h1 + lo*HIDN + kk*32 + hi*8;
        A[kk] = ld16h(base, base + 16);
    }
}

// B: pre-swizzled so each lane's 16 halves are contiguous (32B)
__device__ __forceinline__ h16 loadB(const _Float16* Bsw, int tile, int lane) {
    const _Float16* base = Bsw + (tile*32 + lane)*16;
    return ld16h(base, base + 8);
}

// Swizzle a 128x128 row-major f32 weight matrix into B-fragment layout (f16).
// Tile (kk,j): lane(half,nloc) elem e -> W[kk*32 + ((e&8)?16:0) + half*8 + (e&7)][j*16+nloc]
__device__ __forceinline__ void swizzleB(_Float16 Bsw[2][32][32][16],
                                         const float* src0, const float* src1,
                                         int wave, int lane) {
    const float* srcs[2] = { src0, src1 };
    int half = lane >> 4, nl = lane & 15;
    for (int job = wave; job < 64; job += WAVES) {
        int mlp = job >> 5, tile = job & 31;
        int kk = tile >> 3, j = tile & 7;
        int n = (j << 4) + nl;
#pragma unroll
        for (int e = 0; e < 16; ++e) {
            int K = kk*32 + ((e & 8) ? 16 : 0) + half*8 + (e & 7);
            Bsw[mlp][tile][lane][e] = (_Float16)srcs[mlp][K*HIDN + n];
        }
    }
}

__device__ __forceinline__ float silu(float x) {
    return x * (1.0f / (1.0f + __expf(-x)));
}

// ---------------------------------------------------------------------------

__global__ __launch_bounds__(128, 1)
void VariationalSDE_cdna5_kernel(
    const float* __restrict__ z0,   const float* __restrict__ pctx,
    const float* __restrict__ cctx, const float* __restrict__ times,
    const float* __restrict__ noise,const float* __restrict__ freqs,
    const float* __restrict__ dW1, const float* __restrict__ db1,
    const float* __restrict__ dW2, const float* __restrict__ db2,
    const float* __restrict__ dW3, const float* __restrict__ db3,
    const float* __restrict__ cW1, const float* __restrict__ cb1,
    const float* __restrict__ cW2, const float* __restrict__ cb2,
    const float* __restrict__ cW3, const float* __restrict__ cb3,
    const float* __restrict__ log_diff,
    float* __restrict__ out)
{
    // ---- LDS (~139 KB; CDNA5 WGP has 320 KB) ----
    __shared__ __align__(16) _Float16 s_Bsw[2][32][32][16];        // 64 KB swizzled B (W1ctx, then W2)
    __shared__ __align__(16) _Float16 s_ctxraw[2][WAVES][SPW*HIDN];// 32 KB ctx@W1ctx+b1, raw [m][k] f16
    __shared__ __align__(16) _Float16 s_tembp[2][NST][HIDN];       // 25.6 KB temb@W1temb per step
    __shared__ __align__(16) _Float16 s_h1[WAVES][SPW*HIDN];       // 16 KB staging (ctx) / layer-3 partials
    __shared__ __align__(16) _Float16 s_w1zh[2][2][HIDN];          // 1 KB W1_z rows as f16
    __shared__ float s_z[WAVES][SPW][2];
    __shared__ float s_times[NST + 2];

    const int tid  = threadIdx.x;
    const int wave = tid >> 5;
    const int lane = tid & 31;
    const int lo   = lane & 15;   // WMMA M (A) / N (B,C,D) lane index
    const int hi   = lane >> 4;   // half select
    const int sbase = blockIdx.x * SPB + wave * SPW;

    const size_t TRAJ_N = (size_t)(NST + 1) * Nn * 2;
    const size_t US_N   = (size_t)NST * Nn * 2;

    const float* W1[2]  = { cW1, dW1 };
    const float* B1[2]  = { cb1, db1 };
    const float* W2[2]  = { cW2, dW2 };
    const float* B2[2]  = { cb2, db2 };
    const float* W3[2]  = { cW3, dW3 };
    const float* B3[2]  = { cb3, db3 };
    const float* CTXp[2] = { cctx, pctx };   // mlp0 = f_theta (cnf), mlp1 = u (posterior)

    // ---- phase 1: small shared tables + times output ----
    for (int i = tid; i < HIDN; i += 128) {
#pragma unroll
        for (int m = 0; m < 2; ++m) {
            s_w1zh[m][0][i] = (_Float16)W1[m][0*HIDN + i];
            s_w1zh[m][1][i] = (_Float16)W1[m][1*HIDN + i];
        }
    }
    if (tid < NST + 1) {
        s_times[tid] = times[tid];
        if (blockIdx.x == 0) out[TRAJ_N + US_N + tid] = times[tid];
    }

    // ---- phase 2: swizzle W1_ctx (rows 2..129 of W1) into B buffers ----
    swizzleB(s_Bsw, W1[0] + 2*HIDN, W1[1] + 2*HIDN, wave, lane);

    // ---- phase 3: temb @ W1_temb for all 50 steps (rows 130..161) ----
    for (int pr = tid; pr < 2*NST; pr += 128) {
        int mlp = pr / NST, st = pr % NST;
        float t = times[st];
        float sv[16], cv[16];
#pragma unroll
        for (int q = 0; q < 16; ++q) {
            float a = 6.28318530718f * t * freqs[q];
            sv[q] = __sinf(a); cv[q] = __cosf(a);
        }
        const float* Wt = W1[mlp] + (2 + HIDN)*HIDN;
        for (int n = 0; n < HIDN; ++n) {
            float acc = 0.f;
#pragma unroll
            for (int q = 0; q < 16; ++q)
                acc += sv[q]*Wt[q*HIDN + n] + cv[q]*Wt[(16 + q)*HIDN + n];
            s_tembp[mlp][st][n] = (_Float16)acc;
        }
    }
    __syncthreads();

    // ---- phase 4: ctx_pre = ctx @ W1_ctx + b1 via WMMA; store raw [m][k] ----
    for (int mlp = 0; mlp < 2; ++mlp) {
        // stage ctx tile (16x128) as f16 raw [m][k]; lane covers (m=lo, k half=hi)
        {
            const float* src = CTXp[mlp] + (size_t)(sbase + lo)*HIDN + hi*64;
            _Float16* dst = &s_h1[wave][lo*HIDN + hi*64];
            for (int k = 0; k < 64; ++k) dst[k] = (_Float16)src[k];
        }
        h16 A[4];
        loadA(&s_h1[wave][0], lo, hi, A);
#pragma unroll
        for (int j = 0; j < 8; ++j) {
            int n = (j << 4) + lo;
            f8 acc = { 0.f, 0.f, 0.f, 0.f, 0.f, 0.f, 0.f, 0.f };
#pragma unroll
            for (int kk = 0; kk < 4; ++kk) {
                h16 Bf = loadB(&s_Bsw[mlp][0][0][0], (kk<<3) + j, lane);
                acc = __builtin_amdgcn_wmma_f32_16x16x32_f16(
                          false, A[kk], false, Bf, (short)0, acc, false, false);
            }
            float bv = B1[mlp][n];
            // transpose D-frag -> raw [m][k] (one-time scatter, reused 50 steps)
#pragma unroll
            for (int r = 0; r < 8; ++r)
                s_ctxraw[mlp][wave][(r + (hi << 3))*HIDN + n] = (_Float16)(acc[r] + bv);
        }
    }
    __syncthreads();

    // ---- phase 5: overwrite B buffers with swizzled W2; init z / traj[0] ----
    swizzleB(s_Bsw, W2[0], W2[1], wave, lane);
    {
        int sample = sbase + lo, zd = hi;
        float zv = z0[(size_t)sample*2 + zd];
        s_z[wave][lo][zd] = zv;
        out[(size_t)sample*2 + zd] = zv;     // traj[0]
    }
    __syncthreads();

    // ---- pin ALL layer-2 B fragments in VGPRs (512 regs; gfx1250 has 1024/wave,
    //      and at 139 KB LDS we run 1 workgroup/WGP = 1 wave/SIMD anyway) ----
    h16 Breg[2][32];
#pragma unroll
    for (int mlp = 0; mlp < 2; ++mlp)
#pragma unroll
        for (int t = 0; t < 32; ++t)
            Breg[mlp][t] = loadB(&s_Bsw[mlp][0][0][0], t, lane);

    const float gbase = log1pf(__expf(log_diff[0]));   // softplus * DIFFUSION_SCALE(=1)
    float b3r[2] = { B3[0][hi], B3[1][hi] };
    float b2r[2][8], w3r[2][8][2];                     // per-lane invariants
#pragma unroll
    for (int mlp = 0; mlp < 2; ++mlp)
#pragma unroll
        for (int j = 0; j < 8; ++j) {
            int n = (j << 4) + lo;
            b2r[mlp][j]    = B2[mlp][n];
            w3r[mlp][j][0] = W3[mlp][n*2 + 0];
            w3r[mlp][j][1] = W3[mlp][n*2 + 1];
        }

    float* s_par = (float*)&s_h1[wave][0];   // 2 KB per-wave layer-3 partial buffer

    // ---- phase 6: 50 sequential SDE steps (no cross-wave sync needed) ----
    float fu[2];
    for (int step = 0; step < NST; ++step) {
        float t_i = s_times[step];
        float dt  = s_times[step + 1] - t_i;
        int sample = sbase + lo;

        if (step + 1 < NST)   // prefetch next step's noise -> global_prefetch_b8
            __builtin_prefetch(&noise[((size_t)(step + 1)*Nn + sample)*2 + hi], 0, 1);

#pragma unroll
        for (int mlp = 0; mlp < 2; ++mlp) {
            // ---- layer 1 computed DIRECTLY into A-fragment registers ----
            // A lane role: row m = lo, half = hi; this lane only needs z[lo][*]
            float z0v = s_z[wave][lo][0];
            float z1v = s_z[wave][lo][1];
            h16 A[4];
#pragma unroll
            for (int kk = 0; kk < 4; ++kk) {
                int kbase = kk*32 + hi*8;  // elems 0..7 at k=kbase.., elems 8..15 at k=kbase+16..
                union { uint4 u; _Float16 h[8]; } cp0, cp1, tp0, tp1, w00, w01, w10, w11;
                cp0.u = *(const uint4*)&s_ctxraw[mlp][wave][lo*HIDN + kbase];
                cp1.u = *(const uint4*)&s_ctxraw[mlp][wave][lo*HIDN + kbase + 16];
                tp0.u = *(const uint4*)&s_tembp[mlp][step][kbase];
                tp1.u = *(const uint4*)&s_tembp[mlp][step][kbase + 16];
                w00.u = *(const uint4*)&s_w1zh[mlp][0][kbase];
                w01.u = *(const uint4*)&s_w1zh[mlp][0][kbase + 16];
                w10.u = *(const uint4*)&s_w1zh[mlp][1][kbase];
                w11.u = *(const uint4*)&s_w1zh[mlp][1][kbase + 16];
                union { _Float16 h[16]; h16 v; } a;
#pragma unroll
                for (int e = 0; e < 8; ++e) {
                    float x0 = (float)cp0.h[e] + (float)tp0.h[e]
                             + z0v*(float)w00.h[e] + z1v*(float)w10.h[e];
                    float x1 = (float)cp1.h[e] + (float)tp1.h[e]
                             + z0v*(float)w01.h[e] + z1v*(float)w11.h[e];
                    a.h[e]     = (_Float16)silu(x0);
                    a.h[8 + e] = (_Float16)silu(x1);
                }
                A[kk] = a.v;
            }

            // ---- layer 2 (WMMA, B from registers) fused with layer-3 partial dot ----
            float par[8][2];
#pragma unroll
            for (int r = 0; r < 8; ++r) { par[r][0] = 0.f; par[r][1] = 0.f; }
#pragma unroll
            for (int j = 0; j < 8; ++j) {
                f8 acc = { 0.f, 0.f, 0.f, 0.f, 0.f, 0.f, 0.f, 0.f };
#pragma unroll
                for (int kk = 0; kk < 4; ++kk)
                    acc = __builtin_amdgcn_wmma_f32_16x16x32_f16(
                              false, A[kk], false, Breg[mlp][(kk<<3) + j], (short)0,
                              acc, false, false);
                float bv = b2r[mlp][j];
#pragma unroll
                for (int r = 0; r < 8; ++r) {
                    float s = silu(acc[r] + bv);          // h2[m=r+8hi][n=j*16+lo]
                    par[r][0] += s * w3r[mlp][j][0];
                    par[r][1] += s * w3r[mlp][j][1];
                }
            }

            // ---- layer-3 cross-lane reduction via tiny LDS transpose ----
            {
                float4* dst = (float4*)&s_par[lane*16];
                dst[0] = make_float4(par[0][0], par[0][1], par[1][0], par[1][1]);
                dst[1] = make_float4(par[2][0], par[2][1], par[3][0], par[3][1]);
                dst[2] = make_float4(par[4][0], par[4][1], par[5][0], par[5][1]);
                dst[3] = make_float4(par[6][0], par[6][1], par[7][0], par[7][1]);
            }
            // lane (lo,hi) gathers row m=lo, zd=hi from the 16 lanes of group (lo>>3)
            {
                float sum = b3r[mlp];
                int g = lo >> 3, r = lo & 7;
#pragma unroll
                for (int lp = 0; lp < 16; ++lp)
                    sum += s_par[((g << 4) + lp)*16 + r*2 + hi];
                fu[mlp] = sum;
            }
        }

        // ---- Euler-Maruyama update; lane = (m=lo, zd=hi) ----
        {
            int zd = hi;
            float zc = s_z[wave][lo][zd];
            float xi = noise[((size_t)step*Nn + sample)*2 + zd];
            float g  = gbase * (1.0f - t_i);
            float zn = zc + (fu[0] + fu[1])*dt + xi * (g * sqrtf(fmaxf(dt, 1e-12f)));
            s_z[wave][lo][zd] = zn;
            out[((size_t)(step + 1)*Nn + sample)*2 + zd] = zn;        // traj
            out[TRAJ_N + ((size_t)step*Nn + sample)*2 + zd] = fu[1];  // us
        }
    }
}

extern "C" void kernel_launch(void* const* d_in, const int* in_sizes, int n_in,
                              void* d_out, int out_size, void* d_ws, size_t ws_size,
                              hipStream_t stream) {
    const float* z0    = (const float*)d_in[0];
    const float* pctx  = (const float*)d_in[1];
    const float* cctx  = (const float*)d_in[2];
    const float* times = (const float*)d_in[3];
    const float* noise = (const float*)d_in[4];
    const float* freqs = (const float*)d_in[5];
    const float* dW1 = (const float*)d_in[6];
    const float* db1 = (const float*)d_in[7];
    const float* dW2 = (const float*)d_in[8];
    const float* db2 = (const float*)d_in[9];
    const float* dW3 = (const float*)d_in[10];
    const float* db3 = (const float*)d_in[11];
    const float* cW1 = (const float*)d_in[12];
    const float* cb1 = (const float*)d_in[13];
    const float* cW2 = (const float*)d_in[14];
    const float* cb2 = (const float*)d_in[15];
    const float* cW3 = (const float*)d_in[16];
    const float* cb3 = (const float*)d_in[17];
    const float* ld  = (const float*)d_in[18];

    dim3 grid(Nn / SPB), block(128);
    VariationalSDE_cdna5_kernel<<<grid, block, 0, stream>>>(
        z0, pctx, cctx, times, noise, freqs,
        dW1, db1, dW2, db2, dW3, db3,
        cW1, cb1, cW2, cb2, cW3, cb3, ld,
        (float*)d_out);
}
